// MultiEmbTopModule_26714696581250
// MI455X (gfx1250) — matrix operand order repaired and checked by reading, not compiled, not run on hardware
//
#include <hip/hip_runtime.h>
#include <math.h>

typedef __attribute__((ext_vector_type(16))) __bf16 v16bf;
typedef __attribute__((ext_vector_type(8)))  float  v8f;

// ---- packed f32x2 -> bf16x2 (RNE) ----
#if __has_builtin(__builtin_amdgcn_cvt_pk_bf16_f32)
typedef __attribute__((ext_vector_type(2))) __bf16 v2bf;
static __device__ __forceinline__ unsigned int pk2bf(float x, float y) {
    union { v2bf v; unsigned int u; } r;
    r.v = __builtin_amdgcn_cvt_pk_bf16_f32(x, y);
    return r.u;
}
#else
static __device__ __forceinline__ unsigned int pk2bf(float x, float y) {
    unsigned int ux = __float_as_uint(x);
    unsigned int uy = __float_as_uint(y);
    ux += 0x7fffu + ((ux >> 16) & 1u);        // round-to-nearest-even
    uy += 0x7fffu + ((uy >> 16) & 1u);
    return (ux >> 16) | (uy & 0xffff0000u);
}
#endif

// f32 -> packed bf16 pairs; n must be a multiple of 4 (true for all shapes here)
__global__ void cvt_bf16_kernel(const float* __restrict__ src,
                                unsigned int* __restrict__ dst, int n4)
{
    int i = blockIdx.x * 256 + threadIdx.x;   // one float4 -> uint2 per thread
    if (i >= n4) return;
    float4 v = ((const float4*)src)[i];
    uint2 p;
    p.x = pk2bf(v.x, v.y);
    p.y = pk2bf(v.z, v.w);
    ((uint2*)dst)[i] = p;
}

// ---------------------------------------------------------------------------
// Tiled GEMM:  out[m,n] = sum_k A[m,k] * W[n,k] + bias[n]
// A,W pre-converted to packed bf16 pairs (uint = {K=2j lo16, K=2j+1 hi16}),
// row-major with K fastest (NT = WMMA-native). ldap/ldwp are in dwords (K/2).
// Block tile 32(M) x 128(N), K-step 32; 8 waves in a 2x4 grid, each wave owns
// a 16x32 output strip = two v_wmma_f32_16x16x32_bf16 sharing the A operand.
// All LDS traffic is aligned b128. Caller guarantees M%32==0, N%128==0, K%32==0.
// ---------------------------------------------------------------------------
__global__ __launch_bounds__(256) void gemm_nt_bf16(
    const unsigned int* __restrict__ Apk, int ldap,
    const unsigned int* __restrict__ Wpk, int ldwp,
    const float* __restrict__ bias,
    float* __restrict__ out, int ldo, int K)
{
    // 20-dword row stride: 16B-aligned b128 rows, distinct banks for r=0..15
    __shared__ unsigned int lA[32][20];
    __shared__ unsigned int lW[128][20];

    const int t    = threadIdx.x;
    const int lane = t & 31;
    const int wave = t >> 5;          // 0..7
    const int wm   = wave >> 2;       // 0..1  (M sub-tile)
    const int wn   = wave & 3;        // 0..3  (N sub-strip of 32)
    const int half = lane >> 4;       // 0/1
    const int l16  = lane & 15;

    const int rowBase = blockIdx.y * 32;
    const int colBase = blockIdx.x * 128;

    // cooperative loader: 4 threads per row, one uint4 (8 K-values) each
    const int rr = t >> 2;            // 0..63
    const int c4 = (t & 3) << 2;      // dword col 0,4,8,12

    const unsigned int* Aptr = Apk + (size_t)(rowBase + (rr & 31)) * ldap + c4;
    const unsigned int* Wp0  = Wpk + (size_t)(colBase + rr)      * ldwp + c4;
    const unsigned int* Wp1  = Wpk + (size_t)(colBase + 64 + rr) * ldwp + c4;

    v8f acc0 = {0.f,0.f,0.f,0.f,0.f,0.f,0.f,0.f};
    v8f acc1 = {0.f,0.f,0.f,0.f,0.f,0.f,0.f,0.f};

    const unsigned int* pa  = &lA[wm * 16 + l16][0];
    const unsigned int* pb0 = &lW[wn * 32 + l16][0];
    const unsigned int* pb1 = &lW[wn * 32 + 16 + l16][0];

    const int kdEnd = K >> 1;                  // dwords per row
    for (int kd = 0; kd < kdEnd; kd += 16) {   // 16 dwords = 32 K-values
        __syncthreads();                       // tiles free for overwrite
        if (t < 128) {                         // A: 32 rows x 16 dwords
            uint4 a = *(const uint4*)(Aptr + kd);
            *(uint4*)&lA[rr][c4] = a;
        }
        uint4 w0 = *(const uint4*)(Wp0 + kd);  // W: 128 rows x 16 dwords
        uint4 w1 = *(const uint4*)(Wp1 + kd);
        *(uint4*)&lW[rr][c4]      = w0;
        *(uint4*)&lW[rr + 64][c4] = w1;
        if (kd + 16 < kdEnd) {                 // global_prefetch_b8 next tiles
            __builtin_prefetch(Aptr + kd + 16, 0, 3);
            __builtin_prefetch(Wp0 + kd + 16, 0, 3);
            __builtin_prefetch(Wp1 + kd + 16, 0, 3);
        }
        __syncthreads();

        // Per-lane WMMA operands in contiguous k order (k = 16*half + s, same for A and B) -> contiguous b128 runs:
        //  A 16x32 bf16: dwords {half*8+0..7}          (M = lane%16)
        //  B 32x16 bf16: dwords {half*8+0..7}          (N = lane%16)
        union { uint4 q[2]; v16bf v; } am, b0, b1;
        am.q[0] = *(const uint4*)(pa + (half << 3));
        am.q[1] = *(const uint4*)(pa + (half << 3) + 4);
        b0.q[0] = *(const uint4*)(pb0 + (half << 3));
        b0.q[1] = *(const uint4*)(pb0 + (half << 3) + 4);
        b1.q[0] = *(const uint4*)(pb1 + (half << 3));
        b1.q[1] = *(const uint4*)(pb1 + (half << 3) + 4);

        acc0 = __builtin_amdgcn_wmma_f32_16x16x32_bf16(
            false, am.v, false, b0.v, (short)0, acc0, false, false);
        acc1 = __builtin_amdgcn_wmma_f32_16x16x32_bf16(
            false, am.v, false, b1.v, (short)0, acc1, false, false);
    }

    // D layout: VGPR v -> row (v + 8*half), col lane%16
    const int ocol0 = colBase + wn * 32 + l16;
    const int ocol1 = ocol0 + 16;
    const float bb0 = bias[ocol0];
    const float bb1 = bias[ocol1];
    const int orow0 = rowBase + wm * 16 + half * 8;
#pragma unroll
    for (int vv = 0; vv < 8; ++vv) {
        out[(size_t)(orow0 + vv) * ldo + ocol0] = acc0[vv] + bb0;
        out[(size_t)(orow0 + vv) * ldo + ocol1] = acc1[vv] + bb1;
    }
}

// ---------------------------------------------------------------------------
// MaxPool2d(k,k): act (32,C,H,W) -> x (32*49, C) with x[(b*49+n)*C+c] = max
// ---------------------------------------------------------------------------
__global__ void pool_kernel(const float* __restrict__ act, float* __restrict__ x,
                            int C, int H, int W, int k)
{
    int i = blockIdx.x * 256 + threadIdx.x;      // i = (b*C + c)*49 + n
    if (i >= 32 * C * 49) return;
    int n  = i % 49;
    int bc = i / 49;
    int ph = n / 7, pw = n % 7;
    const float* p = act + ((size_t)bc * H + ph * k) * W + pw * k;
    float m = -3.402823466e38f;
    for (int dy = 0; dy < k; ++dy) {
        const float* pr = p + dy * W;
        for (int dx = 0; dx < k; ++dx) m = fmaxf(m, pr[dx]);
    }
    int b = bc / C, c = bc % C;
    x[((size_t)(b * 49 + n)) * C + c] = m;
}

// xbar[b,c] = mean_n x[b,n,c]
__global__ void xbar_kernel(const float* __restrict__ x, float* __restrict__ xbar, int C)
{
    int i = blockIdx.x * 256 + threadIdx.x;      // b*C + c
    if (i >= 32 * C) return;
    int b = i / C, c = i % C;
    const float* p = x + (size_t)b * 49 * C + c;
    float acc = 0.f;
    for (int n = 0; n < 49; ++n) acc += p[(size_t)n * C];
    xbar[i] = acc * (1.0f / 49.0f);
}

// s[h][a][b*49+m] = (1/sqrt(d)) * dot(Qbar[a, h*d:], K[b*49+m, h*d:])
__global__ void s_kernel(const float* __restrict__ Qbar, const float* __restrict__ Kb,
                         float* __restrict__ s, int C, int d, float scale)
{
    int i = blockIdx.x * 256 + threadIdx.x;      // ((h*32 + a)*1568) + j
    if (i >= 8 * 32 * 1568) return;
    int j = i % 1568;
    int a = (i / 1568) % 32;
    int h = i / (1568 * 32);
    const float4* q  = (const float4*)(Qbar + (size_t)a * C + h * d);
    const float4* kk = (const float4*)(Kb + (size_t)j * C + h * d);
    float acc = 0.f;
    int d4 = d >> 2;
    for (int dd = 0; dd < d4; ++dd) {
        float4 qa = q[dd], kb = kk[dd];
        acc += qa.x * kb.x + qa.y * kb.y + qa.z * kb.z + qa.w * kb.w;
    }
    s[i] = acc * scale;
}

// e[(a*32+b)][c] = sum_m s[h][a][b*49+m] * V[(a*49+m)][c],  h = c/d  (V indexed by a!)
__global__ void e_kernel(const float* __restrict__ s, const float* __restrict__ V,
                         float* __restrict__ e, int C, int d)
{
    int i = blockIdx.x * 256 + threadIdx.x;      // ab*C + c
    if (i >= 1024 * C) return;
    int c  = i % C;
    int ab = i / C;
    int a = ab >> 5, b = ab & 31;
    int h = c / d;
    const float* sp = s + ((size_t)(h * 32 + a)) * 1568 + b * 49;
    const float* vp = V + ((size_t)(a * 49)) * C + c;
    float acc = 0.f;
    for (int m = 0; m < 49; ++m) acc += sp[m] * vp[(size_t)m * C];
    e[i] = acc;
}

// Row-wise L2 normalize: fin (1024 x 512) -> out
__global__ void norm_kernel(const float* __restrict__ fin, float* __restrict__ out)
{
    __shared__ float red[256];
    int row = blockIdx.x;
    int t   = threadIdx.x;
    const float* p = fin + (size_t)row * 512;
    float v0 = p[t], v1 = p[t + 256];
    red[t] = v0 * v0 + v1 * v1;
    __syncthreads();
    for (int sft = 128; sft > 0; sft >>= 1) {
        if (t < sft) red[t] += red[t + sft];
        __syncthreads();
    }
    float inv = 1.0f / sqrtf(red[0]);
    out[(size_t)row * 512 + t]       = v0 * inv;
    out[(size_t)row * 512 + t + 256] = v1 * inv;
}

// ---------------------------------------------------------------------------
extern "C" void kernel_launch(void* const* d_in, const int* in_sizes, int n_in,
                              void* d_out, int out_size, void* d_ws, size_t ws_size,
                              hipStream_t stream)
{
    (void)n_in; (void)out_size; (void)ws_size;
    const float* act[4];
    for (int l = 0; l < 4; ++l) act[l] = (const float*)d_in[l];

    const float *Wq[4], *bq[4], *Wk[4], *bk[4], *Wv[4], *bv[4], *Wo[4], *bo[4];
    const float *Wf = nullptr, *bf = nullptr;

    if (in_sizes[4] == 512 * 3840) {
        // jax tree-flatten (alphabetical): Wf, bf, layer1..4 [Wk,Wo,Wq,Wv,bk,bo,bq,bv]
        Wf = (const float*)d_in[4]; bf = (const float*)d_in[5];
        int idx = 6;
        for (int l = 0; l < 4; ++l, idx += 8) {
            Wk[l] = (const float*)d_in[idx + 0]; Wo[l] = (const float*)d_in[idx + 1];
            Wq[l] = (const float*)d_in[idx + 2]; Wv[l] = (const float*)d_in[idx + 3];
            bk[l] = (const float*)d_in[idx + 4]; bo[l] = (const float*)d_in[idx + 5];
            bq[l] = (const float*)d_in[idx + 6]; bv[l] = (const float*)d_in[idx + 7];
        }
    } else if (in_sizes[5] == in_sizes[4]) {
        // layers first (insertion), alphabetical within layer
        int idx = 4;
        for (int l = 0; l < 4; ++l, idx += 8) {
            Wk[l] = (const float*)d_in[idx + 0]; Wo[l] = (const float*)d_in[idx + 1];
            Wq[l] = (const float*)d_in[idx + 2]; Wv[l] = (const float*)d_in[idx + 3];
            bk[l] = (const float*)d_in[idx + 4]; bo[l] = (const float*)d_in[idx + 5];
            bq[l] = (const float*)d_in[idx + 6]; bv[l] = (const float*)d_in[idx + 7];
        }
        Wf = (const float*)d_in[idx]; bf = (const float*)d_in[idx + 1];
    } else {
        // pure insertion order: per layer Wq,bq,Wk,bk,Wv,bv,Wo,bo; then Wf,bf
        int idx = 4;
        for (int l = 0; l < 4; ++l, idx += 8) {
            Wq[l] = (const float*)d_in[idx + 0]; bq[l] = (const float*)d_in[idx + 1];
            Wk[l] = (const float*)d_in[idx + 2]; bk[l] = (const float*)d_in[idx + 3];
            Wv[l] = (const float*)d_in[idx + 4]; bv[l] = (const float*)d_in[idx + 5];
            Wo[l] = (const float*)d_in[idx + 6]; bo[l] = (const float*)d_in[idx + 7];
        }
        Wf = (const float*)d_in[idx]; bf = (const float*)d_in[idx + 1];
    }

    static const int Cs[4]   = {256, 512, 1024, 2048};
    static const int Hs[4]   = {56, 28, 14, 7};
    static const int Kp[4]   = {8, 4, 2, 1};
    static const int EOFF[4] = {0, 256, 768, 1792};   // concat channel offsets

    // ---- workspace carve-up; f32 region then packed-bf16 region (~120 MB) ----
    float* ws   = (float*)d_ws;
    float* Xb   = ws;                          // 1568*2048 f32
    float* Kb   = Xb   + (size_t)1568 * 2048;
    float* Vb   = Kb   + (size_t)1568 * 2048;
    float* Xbar = Vb   + (size_t)1568 * 2048;  // 32*2048
    float* Qbar = Xbar + (size_t)32 * 2048;
    float* Sb   = Qbar + (size_t)32 * 2048;    // 8*32*1568
    float* Eb   = Sb   + (size_t)8 * 32 * 1568;// 1024*2048
    float* AE   = Eb   + (size_t)1024 * 2048;  // 1024*3840
    float* FIN  = AE   + (size_t)1024 * 3840;  // 1024*512

    unsigned int* pk     = (unsigned int*)(FIN + (size_t)1024 * 512);
    unsigned int* Xpk    = pk;                                // 1568*1024
    unsigned int* Xbarpk = Xpk    + (size_t)1568 * 1024;      // 32*1024
    unsigned int* Epk    = Xbarpk + (size_t)32 * 1024;        // 1024*1024
    unsigned int* AEpk   = Epk    + (size_t)1024 * 1024;      // 1024*1920
    unsigned int* Wfpk   = AEpk   + (size_t)1024 * 1920;      // 512*1920
    unsigned int* Wqpk   = Wfpk   + (size_t)512 * 1920;       // 4x 2048*1024
    unsigned int* Wkpk   = Wqpk   + (size_t)2048 * 1024;
    unsigned int* Wvpk   = Wkpk   + (size_t)2048 * 1024;
    unsigned int* Wopk   = Wvpk   + (size_t)2048 * 1024;

    for (int l = 0; l < 4; ++l) {
        const int C = Cs[l], H = Hs[l], k = Kp[l];
        const int d  = C / 8;
        const int Cd = C >> 1;                 // packed row stride (dwords)
        pool_kernel<<<(32 * C * 49 + 255) / 256, 256, 0, stream>>>(act[l], Xb, C, H, H, k);
        xbar_kernel<<<(32 * C + 255) / 256, 256, 0, stream>>>(Xb, Xbar, C);
        // one-shot bf16 conversions for this layer
        cvt_bf16_kernel<<<(1568 * C / 4 + 255) / 256, 256, 0, stream>>>(Xb, Xpk, 1568 * C / 4);
        cvt_bf16_kernel<<<(32 * C / 4 + 255) / 256, 256, 0, stream>>>(Xbar, Xbarpk, 32 * C / 4);
        cvt_bf16_kernel<<<(C * C / 4 + 255) / 256, 256, 0, stream>>>(Wq[l], Wqpk, C * C / 4);
        cvt_bf16_kernel<<<(C * C / 4 + 255) / 256, 256, 0, stream>>>(Wk[l], Wkpk, C * C / 4);
        cvt_bf16_kernel<<<(C * C / 4 + 255) / 256, 256, 0, stream>>>(Wv[l], Wvpk, C * C / 4);
        cvt_bf16_kernel<<<(C * C / 4 + 255) / 256, 256, 0, stream>>>(Wo[l], Wopk, C * C / 4);
        // Qbar = xbar @ Wq^T + bq            (M=32, N=C, K=C)
        gemm_nt_bf16<<<dim3(C / 128, 1), 256, 0, stream>>>(Xbarpk, Cd, Wqpk, Cd, bq[l], Qbar, C, C);
        // K = x @ Wk^T + bk ; V = x @ Wv^T + bv   (M=1568, N=C, K=C)
        gemm_nt_bf16<<<dim3(C / 128, 49), 256, 0, stream>>>(Xpk, Cd, Wkpk, Cd, bk[l], Kb, C, C);
        gemm_nt_bf16<<<dim3(C / 128, 49), 256, 0, stream>>>(Xpk, Cd, Wvpk, Cd, bv[l], Vb, C, C);
        const float scale = 1.0f / sqrtf((float)d);
        s_kernel<<<(8 * 32 * 1568 + 255) / 256, 256, 0, stream>>>(Qbar, Kb, Sb, C, d, scale);
        e_kernel<<<(1024 * C + 255) / 256, 256, 0, stream>>>(Sb, Vb, Eb, C, d);
        // emb_l = e @ Wo^T + bo -> concat slice of all_emb  (M=1024, N=C, K=C)
        cvt_bf16_kernel<<<(1024 * C / 4 + 255) / 256, 256, 0, stream>>>(Eb, Epk, 1024 * C / 4);
        gemm_nt_bf16<<<dim3(C / 128, 32), 256, 0, stream>>>(Epk, Cd, Wopk, Cd, bo[l],
                                                            AE + EOFF[l], 3840, C);
    }
    // fin = all_emb @ Wf^T + bf   (M=1024, N=512, K=3840)
    cvt_bf16_kernel<<<(1024 * 3840 / 4 + 255) / 256, 256, 0, stream>>>(AE, AEpk, 1024 * 3840 / 4);
    cvt_bf16_kernel<<<(512 * 3840 / 4 + 255) / 256, 256, 0, stream>>>(Wf, Wfpk, 512 * 3840 / 4);
    gemm_nt_bf16<<<dim3(4, 32), 256, 0, stream>>>(AEpk, 1920, Wfpk, 1920, bf, FIN, 512, 3840);
    norm_kernel<<<1024, 256, 0, stream>>>(FIN, (float*)d_out);
}
